// net_mixhop_84524956385827
// MI455X (gfx1250) — compile-verified, matrix operand order checked by reading, not run on hardware
//
#include <hip/hip_runtime.h>

// ---------------------------------------------------------------------------
// MixHop GNN forward on gfx1250 (MI455X).
// Dense GEMMs -> v_wmma_f32_16x16x32_bf16 (one wave per 16x16 output tile).
// Graph propagation -> fp32 atomic segment-sum (feature matrix fits in L2).
// ---------------------------------------------------------------------------

typedef __attribute__((ext_vector_type(16))) __bf16       v16bf;
typedef __attribute__((ext_vector_type(8)))  float        v8f;
typedef __attribute__((ext_vector_type(4)))  unsigned int v4u;

#define N_NODES 50000
#define N_EDGES 800000
#define IN_DIM  128
#define HID     64
#define POWERS  5
#define CONCAT  320   // HID * POWERS
#define OUT_C   40
#define OUT_PAD 64    // OUT padded to WMMA-friendly size (zero padded)
#define FP_PAD  48    // final-proj N padding (3 tiles of 16)

__device__ __forceinline__ float bf2f_lo(unsigned int u) {
  return __uint_as_float(u << 16);
}
__device__ __forceinline__ unsigned short f2bf(float f) {
  unsigned int u = __float_as_uint(f);
  u += 0x7FFFu + ((u >> 16) & 1u);          // round-to-nearest-even
  return (unsigned short)(u >> 16);
}

// --------------------------- WMMA GEMM -------------------------------------
// C[M,N] = A[M,K] * W[N,K]^T (+bias) with A,W bf16 row-major, K contiguous.
// Both fragments use the ISA 16-bit A layout: lane l&15 = row, (l>>4) selects
// K sub-ranges {0..7,16..23} vs {8..15,24..31}; two b128 loads per fragment.
union FragBF { v16bf v; v4u q[2]; };

template <int KT>
__global__ __launch_bounds__(32)
void gemm_bf16_wmma(const unsigned short* __restrict__ A, int lda,
                    const unsigned short* __restrict__ W, int ldb,
                    const float* __restrict__ bias, int nbias,
                    float* __restrict__ Cf, unsigned short* __restrict__ Cb,
                    int ldc, int coloff, int nstore, int relu)
{
  const int tm = blockIdx.x;
  const int tn = blockIdx.y;
  const int l  = threadIdx.x;          // 0..31, full wave (EXEC all ones)
  const int r16   = l & 15;
  const int khalf = (l >> 4) << 3;     // 0 or 8

  const unsigned short* ap = A + (size_t)(tm * 16 + r16) * lda + khalf;
  const unsigned short* bp = W + (size_t)(tn * 16 + r16) * ldb + khalf;

  v8f acc = {0.f, 0.f, 0.f, 0.f, 0.f, 0.f, 0.f, 0.f};
#pragma unroll
  for (int kt = 0; kt < KT; ++kt) {
    FragBF fa, fb;
    fa.q[0] = *(const v4u*)(ap);
    fa.q[1] = *(const v4u*)(ap + 16);
    fb.q[0] = *(const v4u*)(bp);
    fb.q[1] = *(const v4u*)(bp + 16);
    ap += 32; bp += 32;
    acc = __builtin_amdgcn_wmma_f32_16x16x32_bf16(
        /*neg_a=*/false, fa.v, /*neg_b=*/false, fb.v,
        /*c_mod=*/(short)0, acc, /*reuse_a=*/false, /*reuse_b=*/false);
  }

  const int n = tn * 16 + r16;
  float bv = 0.f;
  if (bias != nullptr && n < nbias) bv = bias[n];

  // C layout: VGPR i -> M = i (lanes 0-15) or i+8 (lanes 16-31), N = l&15.
  const int mbase = tm * 16 + ((l >> 4) << 3);
#pragma unroll
  for (int i = 0; i < 8; ++i) {
    float v = acc[i] + bv;
    if (relu) v = fmaxf(v, 0.f);
    if (n < nstore) {
      size_t idx = (size_t)(mbase + i) * ldc + coloff + n;
      if (Cf) Cf[idx] = v;
      if (Cb) Cb[idx] = f2bf(v);
    }
  }
}

// --------------------------- utility kernels -------------------------------
__global__ void zero_f32(float* __restrict__ p, int n) {
  int i = blockIdx.x * blockDim.x + threadIdx.x;
  if (i < n) p[i] = 0.f;
}

__global__ void f32_to_bf16(const float* __restrict__ s,
                            unsigned short* __restrict__ d, int n) {
  int i = blockIdx.x * blockDim.x + threadIdx.x;
  if (i < n) d[i] = f2bf(s[i]);
}

// zero-padded weight pack: src[n,k] f32 -> dst[npad,kpad] bf16
__global__ void pad_w_bf16(const float* __restrict__ s, int n, int k,
                           unsigned short* __restrict__ d, int npad, int kpad) {
  int idx = blockIdx.x * blockDim.x + threadIdx.x;
  if (idx >= npad * kpad) return;
  int i = idx / kpad, j = idx - i * kpad;
  float v = (i < n && j < k) ? s[(size_t)i * k + j] : 0.f;
  d[idx] = f2bf(v);
}

// Pt[n][k] = relu(2 * parsing[k][n]) padded to [64,64]
__global__ void build_pt(const float* __restrict__ parsing,
                         unsigned short* __restrict__ d) {
  int idx = blockIdx.x * blockDim.x + threadIdx.x;
  if (idx >= OUT_PAD * OUT_PAD) return;
  int i = idx >> 6, j = idx & 63;
  float v = 0.f;
  if (i < OUT_C && j < OUT_C) v = fmaxf(2.0f * parsing[j * OUT_C + i], 0.f);
  d[idx] = f2bf(v);
}

// --------------------------- edge weight path ------------------------------
// w[e] = dot64(logits_bf[row[e]], q_bf[col[e]]); reduce sum / sumsq.
__global__ __launch_bounds__(256)
void edge_dot(const int* __restrict__ row, const int* __restrict__ col,
              const unsigned short* __restrict__ Lb,
              const unsigned short* __restrict__ Qb,
              float* __restrict__ w, float* __restrict__ stats, int E)
{
  int e = blockIdx.x * 256 + threadIdx.x;
  float v = 0.f;
  if (e < E) {
    const v4u* s = (const v4u*)(Lb + (size_t)row[e] * OUT_PAD);
    const v4u* q = (const v4u*)(Qb + (size_t)col[e] * OUT_PAD);
#pragma unroll
    for (int i = 0; i < 8; ++i) {     // 8 x b128 = 64 bf16 each side
      v4u su = s[i], qu = q[i];
#pragma unroll
      for (int j = 0; j < 4; ++j) {
        unsigned int a = su[j], b = qu[j];
        v += bf2f_lo(a & 0xFFFFu) * bf2f_lo(b & 0xFFFFu);
        v += __uint_as_float(a & 0xFFFF0000u) * __uint_as_float(b & 0xFFFF0000u);
      }
    }
    w[e] = v;
  }
  __shared__ float rs[256], rq[256];
  int t = threadIdx.x;
  rs[t] = v; rq[t] = v * v;
  __syncthreads();
  for (int s2 = 128; s2 > 0; s2 >>= 1) {
    if (t < s2) { rs[t] += rs[t + s2]; rq[t] += rq[t + s2]; }
    __syncthreads();
  }
  if (t == 0) { atomicAdd(&stats[0], rs[0]); atomicAdd(&stats[1], rq[0]); }
}

__global__ void edge_finalize(float* stats, int E) {
  float mean = stats[0] / (float)E;
  float var  = (stats[1] - (float)E * mean * mean) / (float)(E - 1);
  stats[2] = mean;
  stats[3] = sqrtf(1e-4f / var);
}

__global__ void edge_norm1(const int* __restrict__ row, float* __restrict__ w,
                           float* __restrict__ deg,
                           const float* __restrict__ stats, int E) {
  int e = blockIdx.x * blockDim.x + threadIdx.x;
  if (e >= E) return;
  float wn = (w[e] - stats[2]) * stats[3] + 1.0f;
  w[e] = wn;
  atomicAdd(&deg[row[e]], wn);
}

__global__ void deg_inv(const float* __restrict__ deg,
                        float* __restrict__ dinv, int n) {
  int i = blockIdx.x * blockDim.x + threadIdx.x;
  if (i >= n) return;
  float d = deg[i];
  dinv[i] = (d > 0.f) ? rsqrtf(d) : 0.f;
}

__global__ void edge_norm2(const int* __restrict__ row, const int* __restrict__ col,
                           const float* __restrict__ w,
                           const float* __restrict__ dinv,
                           float* __restrict__ enorm, int E) {
  int e = blockIdx.x * blockDim.x + threadIdx.x;
  if (e >= E) return;
  enorm[e] = w[e] * dinv[row[e]] * dinv[col[e]];
}

// --------------------------- graph propagation -----------------------------
// acc[row[e], k] += enorm[e] * feats_bf16[col[e], k]; 8 columns per thread.
__global__ __launch_bounds__(256)
void prop_scatter(const int* __restrict__ row, const int* __restrict__ col,
                  const float* __restrict__ enorm,
                  const unsigned short* __restrict__ feats,
                  float* __restrict__ acc, int E, int K)
{
  int chunks = K >> 3;
  int total  = E * chunks;
  int idx = blockIdx.x * 256 + threadIdx.x;
  if (idx >= total) return;
  int e  = idx / chunks;
  int kb = (idx - e * chunks) << 3;
  float nw = enorm[e];
  int r = row[e], c = col[e];
  const v4u f = *(const v4u*)(feats + (size_t)c * K + kb);
  float* a = acc + (size_t)r * K + kb;
#pragma unroll
  for (int i = 0; i < 4; ++i) {
    unsigned int u = f[i];
    atomicAdd(a + 2 * i,     nw * bf2f_lo(u & 0xFFFFu));
    atomicAdd(a + 2 * i + 1, nw * __uint_as_float(u & 0xFFFF0000u));
  }
}

// --------------------------- host launch -----------------------------------
extern "C" void kernel_launch(void* const* d_in, const int* in_sizes, int n_in,
                              void* d_out, int out_size, void* d_ws, size_t ws_size,
                              hipStream_t stream)
{
  const float* x       = (const float*)d_in[0];
  const int*   eidx    = (const int*)  d_in[1];
  const float* pm_w1   = (const float*)d_in[2];
  const float* pm_b1   = (const float*)d_in[3];
  const float* pm_w2   = (const float*)d_in[4];
  const float* pm_b2   = (const float*)d_in[5];
  const float* pm_w3   = (const float*)d_in[6];
  const float* pm_b3   = (const float*)d_in[7];
  const float* parsing = (const float*)d_in[8];
  const float* c0w     = (const float*)d_in[9];
  const float* c0b     = (const float*)d_in[10];
  const float* c1w     = (const float*)d_in[11];
  const float* c1b     = (const float*)d_in[12];
  const float* fpw     = (const float*)d_in[13];
  const float* fpb     = (const float*)d_in[14];
  const int* row = eidx;
  const int* col = eidx + N_EDGES;
  float* outp = (float*)d_out;

  // ---- workspace carve-out (~219 MB) ----
  char* ws = (char*)d_ws;
  size_t off = 0;
  auto take = [&](size_t bytes) -> char* {
    char* p = ws + off;
    off = (off + bytes + 255) & ~(size_t)255;
    return p;
  };
  unsigned short* xb      = (unsigned short*)take((size_t)N_NODES * IN_DIM * 2);
  unsigned short* h1b     = (unsigned short*)take((size_t)N_NODES * 512 * 2);
  unsigned short* h2b     = (unsigned short*)take((size_t)N_NODES * HID * 2);
  unsigned short* logitsb = (unsigned short*)take((size_t)N_NODES * OUT_PAD * 2);
  unsigned short* qb      = (unsigned short*)take((size_t)N_NODES * OUT_PAD * 2);
  unsigned short* w1b     = (unsigned short*)take(512 * IN_DIM * 2);
  unsigned short* w2b     = (unsigned short*)take(HID * 512 * 2);
  unsigned short* w3b     = (unsigned short*)take(OUT_PAD * HID * 2);
  unsigned short* ptb     = (unsigned short*)take(OUT_PAD * OUT_PAD * 2);
  unsigned short* c0wb    = (unsigned short*)take((size_t)POWERS * HID * IN_DIM * 2);
  unsigned short* c1wb    = (unsigned short*)take((size_t)POWERS * HID * CONCAT * 2);
  unsigned short* fpwb    = (unsigned short*)take((size_t)FP_PAD * CONCAT * 2);
  float* ew    = (float*)take((size_t)N_EDGES * 4);
  float* enorm = (float*)take((size_t)N_EDGES * 4);
  float* deg   = (float*)take((size_t)N_NODES * 4);
  float* dinv  = (float*)take((size_t)N_NODES * 4);
  float* stats = (float*)take(256);
  unsigned short* fbA    = (unsigned short*)take((size_t)N_NODES * CONCAT * 2);
  unsigned short* fbB    = (unsigned short*)take((size_t)N_NODES * CONCAT * 2);
  unsigned short* h0catb = (unsigned short*)take((size_t)N_NODES * CONCAT * 2);
  unsigned short* h1catb = (unsigned short*)take((size_t)N_NODES * CONCAT * 2);
  // fp32 propagation accumulator (64 MB) overlays dead MLP intermediates
  // (h1b + h2b + logitsb = exactly 64,000,000 B, used only before mixhop).
  float* accf = (float*)h1b;

  const int TPB = 256;
  auto nblk = [](long long n, int tpb) { return (unsigned)((n + tpb - 1) / tpb); };
  const int MT = N_NODES / 16;   // 3125 m-tiles

  // ---- bf16 packs ----
  f32_to_bf16<<<nblk((long long)N_NODES * IN_DIM, TPB), TPB, 0, stream>>>(x, xb, N_NODES * IN_DIM);
  f32_to_bf16<<<nblk(512 * IN_DIM, TPB), TPB, 0, stream>>>(pm_w1, w1b, 512 * IN_DIM);
  f32_to_bf16<<<nblk(HID * 512, TPB), TPB, 0, stream>>>(pm_w2, w2b, HID * 512);
  pad_w_bf16<<<nblk(OUT_PAD * HID, TPB), TPB, 0, stream>>>(pm_w3, OUT_C, HID, w3b, OUT_PAD, HID);
  build_pt<<<nblk(OUT_PAD * OUT_PAD, TPB), TPB, 0, stream>>>(parsing, ptb);
  f32_to_bf16<<<nblk(POWERS * HID * IN_DIM, TPB), TPB, 0, stream>>>(c0w, c0wb, POWERS * HID * IN_DIM);
  f32_to_bf16<<<nblk(POWERS * HID * CONCAT, TPB), TPB, 0, stream>>>(c1w, c1wb, POWERS * HID * CONCAT);
  pad_w_bf16<<<nblk(FP_PAD * CONCAT, TPB), TPB, 0, stream>>>(fpw, OUT_C, CONCAT, fpwb, FP_PAD, CONCAT);

  // ---- pseudo-MLP: 128 -> 512 -> 64 -> 40(pad 64) ----
  gemm_bf16_wmma<4><<<dim3(MT, 32), 32, 0, stream>>>(
      xb, IN_DIM, w1b, IN_DIM, pm_b1, 512, nullptr, h1b, 512, 0, 512, 1);
  gemm_bf16_wmma<16><<<dim3(MT, 4), 32, 0, stream>>>(
      h1b, 512, w2b, 512, pm_b2, HID, nullptr, h2b, HID, 0, HID, 1);
  gemm_bf16_wmma<2><<<dim3(MT, 4), 32, 0, stream>>>(
      h2b, HID, w3b, HID, pm_b3, OUT_C, nullptr, logitsb, OUT_PAD, 0, OUT_PAD, 0);
  // q = logits @ relu(2*parsing)  (Pt = P^T, zero padded)
  gemm_bf16_wmma<2><<<dim3(MT, 4), 32, 0, stream>>>(
      logitsb, OUT_PAD, ptb, OUT_PAD, nullptr, 0, nullptr, qb, OUT_PAD, 0, OUT_PAD, 0);

  // ---- edge weights + GCN normalization ----
  zero_f32<<<nblk(N_NODES, TPB), TPB, 0, stream>>>(deg, N_NODES);
  zero_f32<<<1, 32, 0, stream>>>(stats, 8);
  edge_dot<<<nblk(N_EDGES, TPB), TPB, 0, stream>>>(row, col, logitsb, qb, ew, stats, N_EDGES);
  edge_finalize<<<1, 1, 0, stream>>>(stats, N_EDGES);
  edge_norm1<<<nblk(N_EDGES, TPB), TPB, 0, stream>>>(row, ew, deg, stats, N_EDGES);
  deg_inv<<<nblk(N_NODES, TPB), TPB, 0, stream>>>(deg, dinv, N_NODES);
  edge_norm2<<<nblk(N_EDGES, TPB), TPB, 0, stream>>>(row, col, ew, dinv, enorm, N_EDGES);

  // ---- MixHop layer 0 (K = 128) ----
  const unsigned short* fin = xb;
  unsigned short* fping[2] = {fbA, fbB};
  for (int p = 0; p < POWERS; ++p) {
    gemm_bf16_wmma<4><<<dim3(MT, 4), 32, 0, stream>>>(
        fin, IN_DIM, c0wb + (size_t)p * HID * IN_DIM, IN_DIM,
        c0b + p * HID, HID, nullptr, h0catb, CONCAT, p * HID, HID, 1);
    if (p < POWERS - 1) {
      zero_f32<<<nblk((long long)N_NODES * IN_DIM, TPB), TPB, 0, stream>>>(accf, N_NODES * IN_DIM);
      prop_scatter<<<nblk((long long)N_EDGES * (IN_DIM / 8), TPB), TPB, 0, stream>>>(
          row, col, enorm, fin, accf, N_EDGES, IN_DIM);
      f32_to_bf16<<<nblk((long long)N_NODES * IN_DIM, TPB), TPB, 0, stream>>>(
          accf, fping[p & 1], N_NODES * IN_DIM);
      fin = fping[p & 1];
    }
  }

  // ---- MixHop layer 1 (K = 320) ----
  fin = h0catb;
  for (int p = 0; p < POWERS; ++p) {
    gemm_bf16_wmma<10><<<dim3(MT, 4), 32, 0, stream>>>(
        fin, CONCAT, c1wb + (size_t)p * HID * CONCAT, CONCAT,
        c1b + p * HID, HID, nullptr, h1catb, CONCAT, p * HID, HID, 1);
    if (p < POWERS - 1) {
      zero_f32<<<nblk((long long)N_NODES * CONCAT, TPB), TPB, 0, stream>>>(accf, N_NODES * CONCAT);
      prop_scatter<<<nblk((long long)N_EDGES * (CONCAT / 8), TPB), TPB, 0, stream>>>(
          row, col, enorm, fin, accf, N_EDGES, CONCAT);
      f32_to_bf16<<<nblk((long long)N_NODES * CONCAT, TPB), TPB, 0, stream>>>(
          accf, fping[p & 1], N_NODES * CONCAT);
      fin = fping[p & 1];
    }
  }

  // ---- final projection 320 -> 40, straight into d_out ----
  gemm_bf16_wmma<10><<<dim3(MT, FP_PAD / 16), 32, 0, stream>>>(
      h1catb, CONCAT, fpwb, CONCAT, fpb, OUT_C, outp, nullptr, OUT_C, 0, OUT_C, 0);

  (void)in_sizes; (void)n_in; (void)out_size; (void)ws_size;
}
